// GCNRegressor_35966056137051
// MI455X (gfx1250) — matrix operand order repaired and checked by reading, not compiled, not run on hardware
//
#include <hip/hip_runtime.h>

typedef __attribute__((ext_vector_type(16))) _Float16 v16h;
typedef __attribute__((ext_vector_type(8)))  float    v8f;

#define N_NODES  100000
#define N_EDGES  1250000
#define N_GRAPHS 4096
#define IN_DIM   128
#define HID      64

// ---------------------------------------------------------------- utilities
__global__ void zero_f32(float* __restrict__ p, int n) {
    int i = blockIdx.x * blockDim.x + threadIdx.x;
    if (i < n) p[i] = 0.0f;
}

__global__ void cvt_f32_to_f16(const float* __restrict__ s, _Float16* __restrict__ d, int n) {
    int i = blockIdx.x * blockDim.x + threadIdx.x;
    if (i < n) d[i] = (_Float16)s[i];
}

// ---------------------------------------------------------------- degree / norm
__global__ void deg_kernel(const long long* __restrict__ dst, float* __restrict__ deg, int E) {
    int e = blockIdx.x * blockDim.x + threadIdx.x;
    if (e < E) atomicAdd(&deg[(int)dst[e]], 1.0f);
}

__global__ void dinv_kernel(float* __restrict__ deg, int n) {
    int i = blockIdx.x * blockDim.x + threadIdx.x;
    if (i < n) deg[i] = rsqrtf(deg[i] + 1.0f);   // +1 = self loop; deg >= 1 always
}

// ---------------------------------------------------------------- WMMA GEMM
// Out[M x NOUT] = A[M x K] (f32, cvt to f16 on the fly) @ Wh[K x NOUT] (f16)
// grid.x = M/16, blockDim.x = (NOUT/16)*32, one wave per 16-column tile.
// K, NOUT compile-time -> fully unrolled chain of v_wmma_f32_16x16x32_f16.
template <int K, int NOUT>
__global__ void gemm16_wmma(const float* __restrict__ A,
                            const _Float16* __restrict__ Wh,
                            float* __restrict__ Out) {
    constexpr int NCH = K / 32;                 // K chunks of 32
    const int lane  = threadIdx.x & 31;
    const int wave  = threadIdx.x >> 5;
    const int n0    = wave << 4;
    const int m0    = blockIdx.x << 4;
    const int hi    = lane >> 4;                // 0: lanes 0-15, 1: lanes 16-31
    const int rowA  = m0 + (lane & 15);
    const int khalf = hi << 3;                  // A frag: K offset 0 or 8
    const int koff  = hi << 4;                  // unused; B frag also uses khalf (0 or 8)
    const int col   = n0 + (lane & 15);
    const float* arow = A + (size_t)rowA * K;

    // Preload all B (weight) fragments — M-invariant, WGP$/L2 resident.
    v16h bfrag[NCH];
    #pragma unroll
    for (int kc = 0; kc < NCH; ++kc) {
        #pragma unroll
        for (int e = 0; e < 16; ++e)
            bfrag[kc][e] = Wh[(size_t)(kc * 32 + khalf + e + ((e >> 3) << 3)) * NOUT + col];
    }

    v8f c = {};
    #pragma unroll
    for (int kc = 0; kc < NCH; ++kc) {
        const int kb = kc * 32;
        v16h a;
        #pragma unroll
        for (int j = 0; j < 8; ++j) {
            a[j]     = (_Float16)arow[kb + khalf + j];        // VGPR0-3 run (K=khalf..+7)
            a[j + 8] = (_Float16)arow[kb + khalf + 16 + j];   // VGPR4-7 run (K=khalf+16..+23)
        }
        // (neg_a, A, neg_b, B, c_mod, C, reuse_a, reuse_b)
        c = __builtin_amdgcn_wmma_f32_16x16x32_f16(false, a, false, bfrag[kc],
                                                   (short)0, c, false, false);
    }

    const int rbase = m0 + (hi << 3);
    #pragma unroll
    for (int r = 0; r < 8; ++r)
        Out[(size_t)(rbase + r) * NOUT + col] = c[r];
}

// ---------------------------------------------------------------- edge scatter
// agg[dst] += h[src] * dinv[src]*dinv[dst]   (64 features per edge)
__global__ void edge_scatter(const long long* __restrict__ src,
                             const long long* __restrict__ dst,
                             const float* __restrict__ dinv,
                             const float* __restrict__ h,
                             float* __restrict__ agg, int E) {
    int e = blockIdx.x * blockDim.x + threadIdx.x;
    if (e >= E) return;
    int s = (int)src[e], d = (int)dst[e];
    float w = dinv[s] * dinv[d];
    const float4* hs = (const float4*)(h + (size_t)s * HID);
    float* ad = agg + (size_t)d * HID;
    #pragma unroll
    for (int q = 0; q < HID / 4; ++q) {
        float4 v = hs[q];
        atomicAdd(ad + 4 * q + 0, v.x * w);
        atomicAdd(ad + 4 * q + 1, v.y * w);
        atomicAdd(ad + 4 * q + 2, v.z * w);
        atomicAdd(ad + 4 * q + 3, v.w * w);
    }
}

// ---------------------------------------------------------------- fused self-loop + bias + relu
__global__ void relu_bias_selfloop(const float* __restrict__ agg,
                                   const float* __restrict__ h,
                                   const float* __restrict__ dinv,
                                   const float* __restrict__ bias,
                                   float* __restrict__ out, int total) {
    int t = blockIdx.x * blockDim.x + threadIdx.x;
    if (t >= total) return;
    int n = t >> 6, cidx = t & 63;
    float di = dinv[n];
    float v = agg[t] + h[t] * di * di + bias[cidx];
    out[t] = v > 0.0f ? v : 0.0f;
}

// ---------------------------------------------------------------- pooling
__global__ void pool_sum(const float* __restrict__ hin,
                         const long long* __restrict__ batch,
                         float* __restrict__ sums, int total) {
    int t = blockIdx.x * blockDim.x + threadIdx.x;
    if (t >= total) return;
    int n = t >> 6, cidx = t & 63;
    atomicAdd(&sums[(size_t)batch[n] * HID + cidx], hin[t]);
}

__global__ void pool_cnt(const long long* __restrict__ batch, float* __restrict__ cnt, int n) {
    int i = blockIdx.x * blockDim.x + threadIdx.x;
    if (i < n) atomicAdd(&cnt[(int)batch[i]], 1.0f);
}

__global__ void pool_div(const float* __restrict__ sums, const float* __restrict__ cnt,
                         float* __restrict__ pooled, int total) {
    int t = blockIdx.x * blockDim.x + threadIdx.x;
    if (t >= total) return;
    float c = cnt[t >> 6];
    pooled[t] = sums[t] / (c > 1.0f ? c : 1.0f);
}

// ---------------------------------------------------------------- MLP head tail
// out[g] = relu(t[g,:] + bl1) . Wl2 + bl2
__global__ void head_kernel(const float* __restrict__ t,
                            const float* __restrict__ bl1,
                            const float* __restrict__ Wl2,
                            const float* __restrict__ bl2,
                            float* __restrict__ out, int G) {
    int g = blockIdx.x * blockDim.x + threadIdx.x;
    if (g >= G) return;
    float acc = bl2[0];
    const float* tg = t + (size_t)g * 32;
    #pragma unroll
    for (int j = 0; j < 32; ++j) {
        float v = tg[j] + bl1[j];
        acc += (v > 0.0f ? v : 0.0f) * Wl2[j];
    }
    out[g] = acc;
}

// ---------------------------------------------------------------- launch
extern "C" void kernel_launch(void* const* d_in, const int* in_sizes, int n_in,
                              void* d_out, int out_size, void* d_ws, size_t ws_size,
                              hipStream_t stream) {
    const float*     x    = (const float*)d_in[0];
    const long long* eidx = (const long long*)d_in[1];   // [2, E]
    const long long* src  = eidx;
    const long long* dst  = eidx + N_EDGES;
    const long long* batch = (const long long*)d_in[2];
    const float* W1  = (const float*)d_in[3];  const float* b1  = (const float*)d_in[4];
    const float* W2  = (const float*)d_in[5];  const float* b2  = (const float*)d_in[6];
    const float* W3  = (const float*)d_in[7];  const float* b3  = (const float*)d_in[8];
    const float* Wl1 = (const float*)d_in[9];  const float* bl1 = (const float*)d_in[10];
    const float* Wl2 = (const float*)d_in[11]; const float* bl2 = (const float*)d_in[12];
    float* out = (float*)d_out;

    // workspace layout (floats)
    float* ws    = (float*)d_ws;
    float* dinv  = ws;                               // N
    float* h     = dinv + N_NODES;                   // N*64  (GEMM output)
    float* agg   = h    + (size_t)N_NODES * HID;     // N*64  (scatter target)
    float* hin   = agg  + (size_t)N_NODES * HID;     // N*64  (relu output / next input)
    float* sums  = hin  + (size_t)N_NODES * HID;     // G*64
    float* cnt   = sums + (size_t)N_GRAPHS * HID;    // G
    float* pooled= cnt  + N_GRAPHS;                  // G*64
    float* thead = pooled + (size_t)N_GRAPHS * HID;  // G*32
    _Float16* W1h  = (_Float16*)(thead + (size_t)N_GRAPHS * 32); // 128*64
    _Float16* W2h  = W1h + IN_DIM * HID;             // 64*64
    _Float16* W3h  = W2h + HID * HID;                // 64*64
    _Float16* Wl1h = W3h + HID * HID;                // 64*32

    const int B = 256;
    const int NH = N_NODES * HID;
    const int GH = N_GRAPHS * HID;

    // convert weights to f16 once per call
    cvt_f32_to_f16<<<(IN_DIM*HID + B - 1)/B, B, 0, stream>>>(W1, W1h, IN_DIM*HID);
    cvt_f32_to_f16<<<(HID*HID  + B - 1)/B, B, 0, stream>>>(W2, W2h, HID*HID);
    cvt_f32_to_f16<<<(HID*HID  + B - 1)/B, B, 0, stream>>>(W3, W3h, HID*HID);
    cvt_f32_to_f16<<<(HID*32   + B - 1)/B, B, 0, stream>>>(Wl1, Wl1h, HID*32);

    // degrees (incl. self loop) -> dinv
    zero_f32<<<(N_NODES + B - 1)/B, B, 0, stream>>>(dinv, N_NODES);
    deg_kernel<<<(N_EDGES + B - 1)/B, B, 0, stream>>>(dst, dinv, N_EDGES);
    dinv_kernel<<<(N_NODES + B - 1)/B, B, 0, stream>>>(dinv, N_NODES);

    // ---- layer 1 (K=128) ----
    gemm16_wmma<IN_DIM, HID><<<N_NODES/16, (HID/16)*32, 0, stream>>>(x, W1h, h);
    zero_f32<<<(NH + B - 1)/B, B, 0, stream>>>(agg, NH);
    edge_scatter<<<(N_EDGES + B - 1)/B, B, 0, stream>>>(src, dst, dinv, h, agg, N_EDGES);
    relu_bias_selfloop<<<(NH + B - 1)/B, B, 0, stream>>>(agg, h, dinv, b1, hin, NH);

    // ---- layer 2 (K=64) ----
    gemm16_wmma<HID, HID><<<N_NODES/16, (HID/16)*32, 0, stream>>>(hin, W2h, h);
    zero_f32<<<(NH + B - 1)/B, B, 0, stream>>>(agg, NH);
    edge_scatter<<<(N_EDGES + B - 1)/B, B, 0, stream>>>(src, dst, dinv, h, agg, N_EDGES);
    relu_bias_selfloop<<<(NH + B - 1)/B, B, 0, stream>>>(agg, h, dinv, b2, hin, NH);

    // ---- layer 3 (K=64) ----
    gemm16_wmma<HID, HID><<<N_NODES/16, (HID/16)*32, 0, stream>>>(hin, W3h, h);
    zero_f32<<<(NH + B - 1)/B, B, 0, stream>>>(agg, NH);
    edge_scatter<<<(N_EDGES + B - 1)/B, B, 0, stream>>>(src, dst, dinv, h, agg, N_EDGES);
    relu_bias_selfloop<<<(NH + B - 1)/B, B, 0, stream>>>(agg, h, dinv, b3, hin, NH);

    // ---- mean pool over graphs ----
    zero_f32<<<(GH + N_GRAPHS + B - 1)/B, B, 0, stream>>>(sums, GH + N_GRAPHS); // sums+cnt contiguous
    pool_sum<<<(NH + B - 1)/B, B, 0, stream>>>(hin, batch, sums, NH);
    pool_cnt<<<(N_NODES + B - 1)/B, B, 0, stream>>>(batch, cnt, N_NODES);
    pool_div<<<(GH + B - 1)/B, B, 0, stream>>>(sums, cnt, pooled, GH);

    // ---- MLP head ----
    gemm16_wmma<HID, 32><<<N_GRAPHS/16, (32/16)*32, 0, stream>>>(pooled, Wl1h, thead);
    head_kernel<<<(N_GRAPHS + B - 1)/B, B, 0, stream>>>(thead, bl1, Wl2, bl2, out, N_GRAPHS);
}
